// QuantizedLinear_24721831756576
// MI455X (gfx1250) — matrix operand order repaired and checked by reading, not compiled, not run on hardware
//
#include <hip/hip_runtime.h>
#include <hip/hip_bf16.h>
#include <stdint.h>

// out[M,N] = x[M,K] * W^T[N,K] + bias ; W dequantized from 4-bit (group=8 on K)
// M = 16384, N = 11008, K = 4096.

#define IN_F   4096
#define OUT_F  11008
#define M_TOT  16384
#define BM     128
#define BN     128
#define BK     32
#define KSTEPS (IN_F / BK)   // 128
#define LDB_STRIDE 40        // shorts per B row (32 bf16 + 8 pad) -> 80 B
#define LDA_STRIDE 36        // floats per A row (32 f32 + 4 pad)  -> 144 B

typedef __attribute__((ext_vector_type(16))) __bf16 v16bf;
typedef __attribute__((ext_vector_type(8)))  float  v8f;
typedef __attribute__((ext_vector_type(4)))  unsigned v4u;
typedef __attribute__((ext_vector_type(8)))  int      v8i_t;
typedef __attribute__((ext_vector_type(4)))  int      v4i_t;

union FragB16 { uint4 q[2]; unsigned u[8]; v16bf v; };

#if __has_builtin(__builtin_amdgcn_tensor_load_to_lds) && __has_builtin(__builtin_amdgcn_s_wait_tensorcnt)
#define USE_TDM 1
#else
#define USE_TDM 0
#endif

// Pack the high 16 bits (bf16 truncation) of two f32 bit-patterns into one
// dword: [15:0]=a.hi16 (even K), [31:16]=b.hi16 (odd K). One v_perm_b32.
__device__ __forceinline__ unsigned pk_hi16(unsigned a, unsigned b) {
#if __has_builtin(__builtin_amdgcn_perm)
  // bytes {S0,S1}: idx 0..3 = S1(a), 4..7 = S0(b); sel 0x07060302 -> {b3,b2,a3,a2}
  return __builtin_amdgcn_perm(b, a, 0x07060302u);
#else
  return (a >> 16) | (b & 0xFFFF0000u);
#endif
}

// bf16 pack with round-half-up (cheap: 2 adds + 1 perm).
__device__ __forceinline__ unsigned pk_bf16_rnd(float a, float b) {
  return pk_hi16(__float_as_uint(a) + 0x8000u, __float_as_uint(b) + 0x8000u);
}

#if USE_TDM
// TDM D#: 2D tile (BK x BM) of f32 from x into padded LDS rows.
// pad_interval code 4 = every 32 dwords, pad_amount code 3 = 4 dwords -> 36-dword rows.
__device__ __forceinline__ v8i_t tdm_make_g1() {
  v8i_t g1;
  g1[0] = (2 << 16) | (1 << 20) | (4 << 22) | (3 << 25);   // data_size=4B, pad en, int=32dw, amt=4dw
  g1[1] = (int)((IN_F & 0xFFFF) << 16);                    // tensor_dim0[15:0]=4096
  g1[2] = (int)(((unsigned)IN_F >> 16) | ((M_TOT & 0xFFFF) << 16)); // dim0 hi | dim1 lo
  g1[3] = (int)(((unsigned)M_TOT >> 16) | (BK << 16));     // dim1 hi | tile_dim0=32
  g1[4] = (int)BM;                                         // tile_dim1=128 | tile_dim2=0
  g1[5] = (int)IN_F;                                       // tensor_dim0_stride[31:0]=4096
  g1[6] = 0;
  g1[7] = 0;
  return g1;
}

__device__ __forceinline__ void tdm_load_a(const v8i_t& g1, unsigned lds_byte_off,
                                           const float* gptr) {
  unsigned long long ga = (unsigned long long)(uintptr_t)gptr;
  v4u g0;
  g0[0] = 1u;                                               // count=1 (valid user D#)
  g0[1] = lds_byte_off;                                     // lds_addr
  g0[2] = (unsigned)(ga & 0xFFFFFFFFu);                     // global_addr[31:0]
  g0[3] = (unsigned)((ga >> 32) & 0x1FFFFFFu) | (2u << 30); // global_addr[56:32] | type=2
  v4i_t z4 = {0, 0, 0, 0};
  v8i_t z8 = {0, 0, 0, 0, 0, 0, 0, 0};
  __builtin_amdgcn_tensor_load_to_lds(g0, g1, z4, z4, z8, 0);
}
#endif

__global__ __launch_bounds__(256)
void qlin_wmma_bf16(const float* __restrict__ x,
                    const int*   __restrict__ packed,
                    const float* __restrict__ scales,
                    const float* __restrict__ offsets,
                    const float* __restrict__ bias,
                    float*       __restrict__ out) {
  __shared__ float          lds_a[2][BM * LDA_STRIDE];   // raw f32 x tile (TDM target)
  __shared__ unsigned short lds_b[2][BN * LDB_STRIDE];   // bf16 dequantized W tile

  const int tid   = threadIdx.x;
  const int lane  = tid & 31;
  const int wave  = tid >> 5;
  const int waveM = wave >> 2;       // 0..1 -> 64 rows
  const int waveN = wave & 3;        // 0..3 -> 32 cols
  const int lr    = lane & 15;
  const int lq    = lane >> 4;

  const int nBlock = blockIdx.x * BN;
  const int mBlock = blockIdx.y * BM;

  // ---- cooperative B loader: 2 threads per row, 16 K each ----
  const int row = tid >> 1;
  const int kh  = (tid & 1) * 16;

  const int*   pg = packed  + ((size_t)(nBlock + row) * (IN_F/2) + kh/2);
  const float* sg = scales  + ((size_t)(nBlock + row) * (IN_F/8) + kh/8);
  const float* og = offsets + ((size_t)(nBlock + row) * (IN_F/8) + kh/8);
  const float* xtile = x + (size_t)mBlock * IN_F;        // + ks*BK per stage

#if USE_TDM
  const v8i_t g1c = tdm_make_g1();
#else
  const float* xg = x + ((size_t)(mBlock + row) * IN_F + kh);
  float4 fx0, fx1, fx2, fx3;
#endif

  v8f acc[4][2];
  #pragma unroll
  for (int i = 0; i < 4; ++i)
    #pragma unroll
    for (int j = 0; j < 2; ++j) {
      v8f z = {0.f,0.f,0.f,0.f,0.f,0.f,0.f,0.f};
      acc[i][j] = z;
    }

  int4   pw0, pw1;
  float2 sc, of;

  auto load_b_regs = [&](int ks) {
    const int koff = ks * BK;
    const int4* pp = reinterpret_cast<const int4*>(pg + koff / 2);
    pw0 = pp[0]; pw1 = pp[1];
    sc = *reinterpret_cast<const float2*>(sg + koff / 8);
    of = *reinterpret_cast<const float2*>(og + koff / 8);
  };

  auto store_b_lds = [&](int st) {
    int pv[8] = {pw0.x, pw0.y, pw0.z, pw0.w, pw1.x, pw1.y, pw1.z, pw1.w};
    unsigned pb[8];
    #pragma unroll
    for (int j = 0; j < 8; ++j) {
      const int   b  = pv[j] & 0xFF;
      const float s  = (j < 4) ? sc.x : sc.y;
      const float o  = (j < 4) ? of.x : of.y;
      pb[j] = pk_bf16_rnd((float)(b & 15) * s + o, (float)(b >> 4) * s + o);
    }
    uint4* db = reinterpret_cast<uint4*>(&lds_b[st][row * LDB_STRIDE + kh]);
    db[0] = make_uint4(pb[0], pb[1], pb[2], pb[3]);
    db[1] = make_uint4(pb[4], pb[5], pb[6], pb[7]);
  };

  auto issue_a = [&](int st, int ks) {
#if USE_TDM
    if (wave == 0)
      tdm_load_a(g1c, (unsigned)(uintptr_t)&lds_a[st][0], xtile + ks * BK);
#else
    const float4* src = reinterpret_cast<const float4*>(xg + ks * BK);
    fx0 = src[0]; fx1 = src[1]; fx2 = src[2]; fx3 = src[3];
    float4* dst = reinterpret_cast<float4*>(&lds_a[st][row * LDA_STRIDE + kh]);
    dst[0] = fx0; dst[1] = fx1; dst[2] = fx2; dst[3] = fx3;
#endif
  };

  auto wait_a = [&]() {
#if USE_TDM
    if (wave == 0) __builtin_amdgcn_s_wait_tensorcnt(0);
#endif
  };

  // ---- prologue ----
  issue_a(0, 0);
  load_b_regs(0);
  store_b_lds(0);
  wait_a();
  __syncthreads();

  #pragma unroll 1
  for (int ks = 0; ks < KSTEPS; ++ks) {
    const int cur = ks & 1;
    const int nxt = cur ^ 1;

    if (ks + 1 < KSTEPS) {
      issue_a(nxt, ks + 1);            // TDM DMA overlaps with compute below
      load_b_regs(ks + 1);
      if (ks + 2 < KSTEPS)
        __builtin_prefetch(pg + (size_t)(ks + 2) * (BK / 2), 0, 0);
    }

    // ---- fragments ----
    FragB16 a[4], b[2];
    const uint4* paf = reinterpret_cast<const uint4*>(lds_a[cur]);  // 9 uint4 / row (f32 bits)
    const uint4* pbf = reinterpret_cast<const uint4*>(lds_b[cur]);  // 5 uint4 / row
    #pragma unroll
    for (int mt = 0; mt < 4; ++mt) {
      const int r = waveM * 64 + mt * 16 + lr;
      uint4 f0 = paf[r * 9 + lq * 2];
      uint4 f1 = paf[r * 9 + lq * 2 + 1];
      uint4 f2 = paf[r * 9 + 4 + lq * 2];
      uint4 f3 = paf[r * 9 + 4 + lq * 2 + 1];
      a[mt].u[0] = pk_hi16(f0.x, f0.y); a[mt].u[1] = pk_hi16(f0.z, f0.w);
      a[mt].u[2] = pk_hi16(f1.x, f1.y); a[mt].u[3] = pk_hi16(f1.z, f1.w);
      a[mt].u[4] = pk_hi16(f2.x, f2.y); a[mt].u[5] = pk_hi16(f2.z, f2.w);
      a[mt].u[6] = pk_hi16(f3.x, f3.y); a[mt].u[7] = pk_hi16(f3.z, f3.w);
    }
    #pragma unroll
    for (int nt = 0; nt < 2; ++nt) {
      const int c = waveN * 32 + nt * 16 + lr;
      b[nt].q[0] = pbf[c * 5 + lq];
      b[nt].q[1] = pbf[c * 5 + 2 + lq];
    }

    #pragma unroll
    for (int mt = 0; mt < 4; ++mt)
      #pragma unroll
      for (int nt = 0; nt < 2; ++nt)
        acc[mt][nt] = __builtin_amdgcn_wmma_f32_16x16x32_bf16(
            false, a[mt].v, false, b[nt].v, (short)0, acc[mt][nt], false, false);

    if (ks + 1 < KSTEPS) store_b_lds(nxt);
    wait_a();
    __syncthreads();
  }

  // ---- epilogue: bias + store ----
  #pragma unroll
  for (int mt = 0; mt < 4; ++mt) {
    #pragma unroll
    for (int nt = 0; nt < 2; ++nt) {
      const int nn = nBlock + waveN * 32 + nt * 16 + lr;
      const int mm = mBlock + waveM * 64 + mt * 16 + lq * 8;
      const float bv = bias[nn];
      float* op = out + (size_t)mm * OUT_F + nn;
      #pragma unroll
      for (int r2 = 0; r2 < 8; ++r2)
        op[(size_t)r2 * OUT_F] = acc[mt][nt][r2] + bv;
    }
  }
}

extern "C" void kernel_launch(void* const* d_in, const int* in_sizes, int n_in,
                              void* d_out, int out_size, void* d_ws, size_t ws_size,
                              hipStream_t stream) {
  const float* x       = (const float*)d_in[0];
  const int*   packed  = (const int*)  d_in[1];
  const float* scales  = (const float*)d_in[2];
  const float* offsets = (const float*)d_in[3];
  const float* bias    = (const float*)d_in[4];
  float* out = (float*)d_out;

  dim3 grid(OUT_F / BN, M_TOT / BM);   // (86, 128)
  dim3 block(256);
  hipLaunchKernelGGL(qlin_wmma_bf16, grid, block, 0, stream,
                     x, packed, scales, offsets, bias, out);
}
